// PSAResNet_72533407695361
// MI455X (gfx1250) — matrix operand order repaired and checked by reading, not compiled, not run on hardware
//
#include <hip/hip_runtime.h>
#include <hip/hip_bf16.h>

// ---------------------------------------------------------------------------
// PSA attention block for MI455X (gfx1250, wave32, WMMA bf16 16x16x32).
// All matmuls lowered to v_wmma_f32_16x16x32_bf16; fp32 accumulate.
// Working set (~130MB) fits in 192MB L2 -> register-blocked GEMM reading
// fragments directly from global/L2 (explicit addrspace(1) so the loop-carried
// fragment pointers lower to global_load_b128, not flat_load).
// ---------------------------------------------------------------------------

typedef __bf16 bf16_t;
typedef float  v8f   __attribute__((ext_vector_type(8)));
typedef bf16_t v16bf __attribute__((ext_vector_type(16)));
typedef bf16_t v8bf  __attribute__((ext_vector_type(8)));

#define GAS __attribute__((address_space(1)))
typedef GAS const bf16_t* gcbf;
typedef GAS bf16_t*       gbf;
typedef GAS const v8bf*   gcv8;
typedef GAS float*        gf32;
typedef GAS const float*  gcf32;

union BF16x16 { v16bf v; v8bf h[2]; };

// A-matrix 16x32 bf16 fragment (ISA 7.12.2): lane<16 -> row M=lane, K
// {0..7,16..23}; lane>=16 -> same row, K {8..15,24..31}. Caller folds the
// (lane>>4)*8 element phase into p; chunks live at p and p+16.
__device__ __forceinline__ v16bf load_fragA(gcbf p) {
  BF16x16 u;
  u.h[0] = *(gcv8)(p);
  u.h[1] = *(gcv8)(p + 16);
  return u.v;
}

// B-matrix 32x16 bf16 fragment: lane<16 -> col N=lane, K {0..7,16..23};
// lane>=16 -> same col, K {8..15,24..31}. Caller folds the (lane>>4)*8
// element phase into p; chunks live at p and p+16.
__device__ __forceinline__ v16bf load_fragB(gcbf p) {
  BF16x16 u;
  u.h[0] = *(gcv8)(p);
  u.h[1] = *(gcv8)(p + 16);
  return u.v;
}

// Canonical GEMM: D[b][m][n] = epi( sum_k A[b][m][k]*Bt[b][n][k] + bias )
// EPI:  0 none, 1 relu, 2 sigmoid.   BIAS: 0 none, 1 bias[m], 2 bias[n].
template <int EPI, int BIAS, bool OUT_BF16>
__global__ __launch_bounds__(128) void gemm_wmma(
    const bf16_t* __restrict__ A, long lda, long bsA,
    const bf16_t* __restrict__ Bt, long ldb, long bsB,
    void* __restrict__ D, long ldd, long bsD,
    const float* __restrict__ bias, int K) {
  const int lane = threadIdx.x & 31;
  const int wave = threadIdx.x >> 5;
  const long blockM = (long)blockIdx.y * 128 + (wave >> 1) * 64;
  const long blockN = (long)blockIdx.x * 128 + (wave & 1) * 64;
  const long b = blockIdx.z;

  const int rsel = lane & 15;          // row (A) / col (B) within 16-wide tile
  const int kselA = (lane >> 4) * 8;   // A k-phase (interleaved halves)
  const int kselB = (lane >> 4) * 8;   // B k-phase (interleaved halves)

  const gcbf Ag = (gcbf)(A + b * bsA);
  const gcbf Bg = (gcbf)(Bt + b * bsB);

  // Per-lane fragment base pointers (global AS), advanced by 32 elements per
  // k-step so the compiler uses immediate offsets + simple pointer bumps.
  gcbf pa[4];
  gcbf pb[4];
#pragma unroll
  for (int mi = 0; mi < 4; ++mi)
    pa[mi] = Ag + (blockM + mi * 16 + rsel) * lda + kselA;
#pragma unroll
  for (int ni = 0; ni < 4; ++ni)
    pb[ni] = Bg + (blockN + ni * 16 + rsel) * ldb + kselB;

  v8f acc[4][4];
#pragma unroll
  for (int i = 0; i < 4; ++i)
#pragma unroll
    for (int j = 0; j < 4; ++j)
      acc[i][j] = v8f{0.f, 0.f, 0.f, 0.f, 0.f, 0.f, 0.f, 0.f};

#pragma unroll 2
  for (int k0 = 0; k0 < K; k0 += 32) {
    v16bf af[4], bfx[4];
#pragma unroll
    for (int mi = 0; mi < 4; ++mi) { af[mi] = load_fragA(pa[mi]); pa[mi] += 32; }
#pragma unroll
    for (int ni = 0; ni < 4; ++ni) { bfx[ni] = load_fragB(pb[ni]); pb[ni] += 32; }
#pragma unroll
    for (int mi = 0; mi < 4; ++mi)
#pragma unroll
      for (int ni = 0; ni < 4; ++ni)
        acc[mi][ni] = __builtin_amdgcn_wmma_f32_16x16x32_bf16(
            false, af[mi], false, bfx[ni], (short)0, acc[mi][ni], false, false);
  }

  // D layout: VGPR r, lane<16 -> (M=r, N=lane); lane>=16 -> (M=r+8, N=lane-16)
  const int mphase = (lane >> 4) * 8;
  const int ncol = lane & 15;
  const gf32 Df = (gf32)((float*)D + b * bsD);
  const gbf  Dh = (gbf)((bf16_t*)D + b * bsD);
  const gcf32 biasg = (gcf32)bias;
#pragma unroll
  for (int mi = 0; mi < 4; ++mi) {
#pragma unroll
    for (int ni = 0; ni < 4; ++ni) {
#pragma unroll
      for (int r = 0; r < 8; ++r) {
        const long m = blockM + mi * 16 + mphase + r;
        const long n = blockN + ni * 16 + ncol;
        float v = acc[mi][ni][r];
        if (BIAS == 1) v += biasg[m];
        if (BIAS == 2) v += biasg[n];
        if (EPI == 1) v = v > 0.f ? v : 0.f;
        if (EPI == 2) v = 1.f / (1.f + __expf(-v));
        const long idx = m * ldd + n;
        if (OUT_BF16) Dh[idx] = (bf16_t)v;
        else          Df[idx] = v;
      }
    }
  }
}

// x[B,C,H,W] fp32 -> pixel-major bf16 xt[n][c] and masked mxt[n][c]
__global__ __launch_bounds__(256) void prep_x(
    const float* __restrict__ x, const unsigned char* __restrict__ mask,
    bf16_t* __restrict__ xt, bf16_t* __restrict__ mxt) {
  const long idx = (long)blockIdx.x * 256 + threadIdx.x;  // over 4096*2048
  const int n = (int)(idx >> 11);
  const int c = (int)(idx & 2047);
  const int b = n >> 10, p = n & 1023;
  const float v = x[((long)b * 2048 + c) * 1024 + p];
  xt[idx] = (bf16_t)v;
  mxt[idx] = (bf16_t)(mask[n] ? v : 0.0f);
}

__global__ __launch_bounds__(256) void f32_to_bf16(
    const float* __restrict__ src, bf16_t* __restrict__ dst, long n) {
  const long i = (long)blockIdx.x * 256 + threadIdx.x;
  if (i < n) dst[i] = (bf16_t)src[i];
}

// Row softmax: S fp32 [rows,1024] -> attn bf16, one 256-thread block per row.
__global__ __launch_bounds__(256) void softmax_rows(
    const float* __restrict__ S, bf16_t* __restrict__ attn, int ncols) {
  __shared__ float red[8];
  const long row = blockIdx.x;
  const float* s = S + row * ncols;
  bf16_t* a = attn + row * ncols;
  float lmax = -3.4e38f;
  for (int i = threadIdx.x; i < ncols; i += 256) lmax = fmaxf(lmax, s[i]);
  for (int o = 16; o > 0; o >>= 1) lmax = fmaxf(lmax, __shfl_xor(lmax, o, 32));
  if ((threadIdx.x & 31) == 0) red[threadIdx.x >> 5] = lmax;
  __syncthreads();
  float m = red[0];
#pragma unroll
  for (int w = 1; w < 8; ++w) m = fmaxf(m, red[w]);
  __syncthreads();
  float lsum = 0.f;
  for (int i = threadIdx.x; i < ncols; i += 256) lsum += __expf(s[i] - m);
  for (int o = 16; o > 0; o >>= 1) lsum += __shfl_xor(lsum, o, 32);
  if ((threadIdx.x & 31) == 0) red[threadIdx.x >> 5] = lsum;
  __syncthreads();
  float sum = 0.f;
#pragma unroll
  for (int w = 0; w < 8; ++w) sum += red[w];
  const float inv = 1.f / sum;
  for (int i = threadIdx.x; i < ncols; i += 256)
    a[i] = (bf16_t)(__expf(s[i] - m) * inv);
}

// gs[n] = sigmoid(W2 . Ht[n][:] + b2), one wave32 per pixel.
__global__ __launch_bounds__(256) void gate_strength_k(
    const bf16_t* __restrict__ Ht, const float* __restrict__ W2,
    const float* __restrict__ b2, float* __restrict__ gs) {
  const int pix = blockIdx.x * 8 + (threadIdx.x >> 5);
  const int lane = threadIdx.x & 31;
  const bf16_t* h = Ht + (long)pix * 256;
  float acc = 0.f;
#pragma unroll
  for (int i = 0; i < 8; ++i) acc += (float)h[lane + 32 * i] * W2[lane + 32 * i];
  for (int o = 16; o > 0; o >>= 1) acc += __shfl_xor(acc, o, 32);
  if (lane == 0) gs[pix] = 1.f / (1.f + __expf(-(acc + b2[0])));
}

// out[b,c,p] = flag[b] ? out_t[n][c]*Gt[n][c]*gs[n] : x[b,c,p]
__global__ __launch_bounds__(256) void final_combine(
    const float* __restrict__ x, const bf16_t* __restrict__ out_t,
    const bf16_t* __restrict__ Gt, const float* __restrict__ gs,
    const unsigned char* __restrict__ flag, float* __restrict__ out) {
  const long idx = (long)blockIdx.x * 256 + threadIdx.x;  // over 4*2048*1024
  const int p = (int)(idx & 1023);
  const int c = (int)((idx >> 10) & 2047);
  const int b = (int)(idx >> 21);
  if (flag[b]) {
    const long n = (long)b * 1024 + p;
    const long t = n * 2048 + c;
    out[idx] = (float)out_t[t] * (float)Gt[t] * gs[n];
  } else {
    out[idx] = x[idx];
  }
}

extern "C" void kernel_launch(void* const* d_in, const int* in_sizes, int n_in,
                              void* d_out, int out_size, void* d_ws, size_t ws_size,
                              hipStream_t stream) {
  const float* x  = (const float*)d_in[0];
  const float* Wq = (const float*)d_in[1];  const float* bq = (const float*)d_in[2];
  const float* Wk = (const float*)d_in[3];  const float* bk = (const float*)d_in[4];
  const float* Wv = (const float*)d_in[5];  const float* bv = (const float*)d_in[6];
  const float* Wg = (const float*)d_in[7];  const float* bg = (const float*)d_in[8];
  const float* W1 = (const float*)d_in[9];  const float* b1 = (const float*)d_in[10];
  const float* W2 = (const float*)d_in[11]; const float* b2 = (const float*)d_in[12];
  const unsigned char* mask = (const unsigned char*)d_in[13];
  const unsigned char* ueg  = (const unsigned char*)d_in[14];

  char* ws = (char*)d_ws;
  const size_t MB = 1024 * 1024;
  bf16_t* xt   = (bf16_t*)(ws + 0 * MB);    // [4096,2048] bf16  (16 MB)
  bf16_t* mxt  = (bf16_t*)(ws + 16 * MB);   // [4096,2048] bf16  (16 MB)
  bf16_t* Wq_b = (bf16_t*)(ws + 32 * MB);   // 8 MB each
  bf16_t* Wk_b = (bf16_t*)(ws + 40 * MB);
  bf16_t* Wv_b = (bf16_t*)(ws + 48 * MB);
  bf16_t* Wg_b = (bf16_t*)(ws + 56 * MB);
  bf16_t* W1_b = (bf16_t*)(ws + 64 * MB);   // [256,2048] bf16 (1 MB)
  bf16_t* Qt   = (bf16_t*)(ws + 65 * MB);   // [4096,2048] (16 MB)
  bf16_t* Kt   = (bf16_t*)(ws + 81 * MB);   // [4096,2048] (16 MB)
  bf16_t* V    = (bf16_t*)(ws + 97 * MB);   // [2048,4096] (16 MB)
  bf16_t* Gt   = (bf16_t*)(ws + 113 * MB);  // [4096,2048] (16 MB)
  bf16_t* Ht   = (bf16_t*)(ws + 129 * MB);  // [4096,256]  (2 MB)
  float*  gs   = (float*)(ws + 131 * MB);   // [4096] (16 KB)
  // Ordering-safe aliases (all producers of the aliased region finish first):
  float*  S     = (float*)mxt;   // [4,1024,1024] fp32, after Q/K projections
  bf16_t* attn  = (bf16_t*)xt;   // [4,1024,1024] bf16, after all projections
  bf16_t* out_t = (bf16_t*)Qt;   // [4096,2048]  bf16, after S uses Qt

  // 1) pack/transpose x -> bf16 (plus masked copy)
  prep_x<<<32768, 256, 0, stream>>>(x, mask, xt, mxt);
  // 2) weights fp32 -> bf16
  f32_to_bf16<<<16384, 256, 0, stream>>>(Wq, Wq_b, 2048L * 2048);
  f32_to_bf16<<<16384, 256, 0, stream>>>(Wk, Wk_b, 2048L * 2048);
  f32_to_bf16<<<16384, 256, 0, stream>>>(Wv, Wv_b, 2048L * 2048);
  f32_to_bf16<<<16384, 256, 0, stream>>>(Wg, Wg_b, 2048L * 2048);
  f32_to_bf16<<<2048, 256, 0, stream>>>(W1, W1_b, 256L * 2048);

  // 3) projections (pre-transposed where the attention GEMMs want it)
  // Qt[n][o] = mxt[n][:] . Wq[o][:] + bq[o]
  gemm_wmma<0, 2, true><<<dim3(16, 32, 1), 128, 0, stream>>>(
      mxt, 2048, 0, Wq_b, 2048, 0, Qt, 2048, 0, bq, 2048);
  gemm_wmma<0, 2, true><<<dim3(16, 32, 1), 128, 0, stream>>>(
      mxt, 2048, 0, Wk_b, 2048, 0, Kt, 2048, 0, bk, 2048);
  // V[o][n] = Wv[o][:] . xt[n][:] + bv[o]   (channel-major)
  gemm_wmma<0, 1, true><<<dim3(32, 16, 1), 128, 0, stream>>>(
      Wv_b, 2048, 0, xt, 2048, 0, V, 4096, 0, bv, 2048);
  // Gt[n][o] = sigmoid(xt[n][:] . Wg[o][:] + bg[o])
  gemm_wmma<2, 2, true><<<dim3(16, 32, 1), 128, 0, stream>>>(
      xt, 2048, 0, Wg_b, 2048, 0, Gt, 2048, 0, bg, 2048);
  // Ht[n][h] = relu(xt[n][:] . W1[h][:] + b1[h])
  gemm_wmma<1, 2, true><<<dim3(2, 32, 1), 128, 0, stream>>>(
      xt, 2048, 0, W1_b, 2048, 0, Ht, 256, 0, b1, 2048);
  // gate strength head (tiny)
  gate_strength_k<<<512, 256, 0, stream>>>(Ht, W2, b2, gs);

  // 4) attention scores per batch: S[b][q][k] = Qt[b,q,:] . Kt[b,k,:]
  gemm_wmma<0, 0, false><<<dim3(8, 8, 4), 128, 0, stream>>>(
      Qt, 2048, 1024L * 2048, Kt, 2048, 1024L * 2048,
      S, 1024, 1024L * 1024, nullptr, 2048);
  // 5) softmax over k, emit bf16 attn
  softmax_rows<<<4096, 256, 0, stream>>>(S, attn, 1024);
  // 6) out_t[b][m][c] = sum_n attn[b,m,n] * V[c, b*1024+n]
  gemm_wmma<0, 0, true><<<dim3(16, 8, 4), 128, 0, stream>>>(
      attn, 1024, 1024L * 1024, V, 4096, 1024,
      out_t, 2048, 1024L * 2048, nullptr, 1024);

  // 7) gated combine / passthrough
  final_combine<<<32768, 256, 0, stream>>>(x, out_t, Gt, gs, ueg,
                                           (float*)d_out);
}
